// TempSpatialModule_1511828488299
// MI455X (gfx1250) — compile-verified, matrix operand order checked by reading, not run on hardware
//
#include <hip/hip_runtime.h>

#define HW   36864      // 192*192
#define W_   192
#define CIN  128
#define NQKV 384        // 128 q + 128 k + 128 v
#define BATCH 4

typedef __attribute__((ext_vector_type(16))) __bf16 v16bf;
typedef __attribute__((ext_vector_type(8)))  __bf16 v8bf;
typedef __attribute__((ext_vector_type(8)))  float  v8f;
typedef __attribute__((ext_vector_type(4)))  float  v4f;

__device__ __forceinline__ __bf16 f2bf(float f) { return (__bf16)f; }
__device__ __forceinline__ float  bf2f(__bf16 b) { return (float)b; }

// Inverse of the 16-bit WMMA fragment K-mapping (ISA 7.12.2):
//   fragment element h of lane-group g holds K = kk, where
//   g = (kk>>3)&1 ; r = kk-8g ; h = r<8 ? r : r-8
__device__ __forceinline__ void fragpos(int kk, int& g, int& h) {
  g = (kk >> 3) & 1;
  int r = kk - (g << 3);
  h = (r < 8) ? r : (r - 8);
}

// ---------------------------------------------------------------------------
// Kernel 1: fused q = x*q_w^T + q_b ; kv = hx*kv_w^T  -> bf16 qkv[p][384]
// Block: 256 thr (8 waves), M-tile 32 pixels, N 384, K 128 (staged by 32).
// All WMMA operands pre-swizzled in LDS to fragment layout [tile][lane][16].
// ---------------------------------------------------------------------------
__global__ __launch_bounds__(256)
void qkv_kernel(const float* __restrict__ x, const float* __restrict__ hx,
                const float* __restrict__ q_w, const float* __restrict__ q_b,
                const float* __restrict__ kv_w, __bf16* __restrict__ qkv)
{
  __shared__ __attribute__((aligned(32))) __bf16 sXf[4][2][32][16]; // 8 KB
  __shared__ __attribute__((aligned(32))) __bf16 sHf[4][2][32][16]; // 8 KB
  __shared__ __attribute__((aligned(32))) __bf16 sWf[24][32][16];   // 24 KB (reused as out-stage)
  __shared__ float sBias[NQKV];

  const int t  = threadIdx.x;
  const int p0 = blockIdx.x * 32;           // 32 consecutive pixels, same image
  const int b  = p0 / HW;
  const long xbase = (long)b * CIN * HW + (p0 % HW);

  // stage inputs (coalesced over pixels), swizzled to A-fragment layout
  for (int idx = t; idx < 32 * CIN; idx += 256) {
    int m = idx & 31, c = idx >> 5;
    int ks = c >> 5, kk = c & 31, g, h;
    fragpos(kk, g, h);
    int lane = (g << 4) | (m & 15);
    int mt = m >> 4;
    sXf[ks][mt][lane][h] = f2bf(x [xbase + (long)c * HW + m]);
    sHf[ks][mt][lane][h] = f2bf(hx[xbase + (long)c * HW + m]);
  }
  for (int n = t; n < NQKV; n += 256)
    sBias[n] = (n < CIN) ? q_b[n] : 0.0f;

  const int wv = t >> 5, ln = t & 31;
  const int lm = ln & 15, lg = ln >> 4;

  v8f zero = {};
  v8f acc[6];
  #pragma unroll
  for (int i = 0; i < 6; ++i) acc[i] = zero;

  for (int ks = 0; ks < 4; ++ks) {
    const int k0 = ks * 32;
    __syncthreads();
    // stage one K-chunk of combined weights, B-fragment layout (coalesced reads)
    for (int idx = t; idx < 32 * NQKV; idx += 256) {
      int kk = idx & 31, n = idx >> 5, g, h;
      float wval = (n < CIN) ? q_w[n * CIN + (k0 + kk)]
                             : kv_w[(n - CIN) * CIN + (k0 + kk)];
      fragpos(kk, g, h);
      sWf[n >> 4][(g << 4) | (n & 15)][h] = f2bf(wval);
    }
    __syncthreads();

    #pragma unroll
    for (int i = 0; i < 6; ++i) {
      const int tt = wv * 6 + i;              // 48 tiles: 2 M x 24 N
      const int nt = tt % 24, mt = tt / 24;
      v16bf A  = (nt < 8) ? *(const v16bf*)&sXf[ks][mt][ln][0]
                          : *(const v16bf*)&sHf[ks][mt][ln][0];
      v16bf Bf = *(const v16bf*)&sWf[nt][ln][0];
      acc[i] = __builtin_amdgcn_wmma_f32_16x16x32_bf16(
          false, A, false, Bf, (short)0, acc[i], false, false);
    }
  }

  // stage D tiles into LDS (alias sWf as bf16 [32][384]), then wide store
  __syncthreads();
  __bf16* sOut = &sWf[0][0][0];               // 32*384 elements exactly
  #pragma unroll
  for (int i = 0; i < 6; ++i) {
    const int tt = wv * 6 + i;
    const int nt = tt % 24, mt = tt / 24;
    const int n = nt * 16 + lm;
    const float bias = sBias[n];
    #pragma unroll
    for (int r = 0; r < 8; ++r) {
      const int m = mt * 16 + r + (lg << 3);
      sOut[m * NQKV + n] = f2bf(acc[i][r] + bias);
    }
  }
  __syncthreads();
  {
    const uint4* s4 = (const uint4*)sOut;
    uint4* d4 = (uint4*)(qkv + (long)p0 * NQKV);
    #pragma unroll
    for (int idx = t; idx < (32 * NQKV) / 8; idx += 256)   // 6 iters, coalesced
      d4[idx] = s4[idx];
  }
}

// ---------------------------------------------------------------------------
// Kernel 2: dilated window attention (+torch reshape scramble) + residual
//           + LayerNorm + proj GEMM (+bias +residual) -> out [B,C,H,W] f32
// Block: 128 thr (4 waves), 64 output pixels.
// ---------------------------------------------------------------------------
__global__ __launch_bounds__(128)
void attn_ln_proj_kernel(const __bf16* __restrict__ qkv,
                         const float* __restrict__ x,
                         const float* __restrict__ ln_g,
                         const float* __restrict__ ln_b,
                         const float* __restrict__ proj_w,
                         const float* __restrict__ proj_b,
                         float* __restrict__ out)
{
  __shared__ float sXR[64][CIN + 2];                                // 33 KB (padded)
  __shared__ __attribute__((aligned(32))) __bf16 sXN[4][4][32][16]; // 16 KB (A frags)
  __shared__ __attribute__((aligned(32))) __bf16 sPWf[8][32][16];   // 8 KB  (B frag chunk)
  __shared__ float sMu[64], sRs[64];

  const int t  = threadIdx.x;
  const int p0 = blockIdx.x * 64;

  // -------- phase 1: attention, one (pixel, half) per thread --------
  {
    const int pix  = t >> 1;
    const int half = t & 1;
    const int pp = p0 + pix;
    const long T = 2L * pp + half;                 // torch stack+reshape quirk
    const int f  = (int)(T / (long)(BATCH * HW));  // frame (0: dil 1, 1: dil 2)
    const int s  = (int)(T % (long)(BATCH * HW));  // source pixel index
    const int bs = s / HW, qs = s % HW;
    const int hs = qs / W_, wsrc = qs % W_;
    const int dil = 1 << f;
    const int bp = pp / HW, qp = pp % HW;
    const long xres = (long)bp * CIN * HW + qp;
    const float scale = 0.17677669529663687f;      // 32^-0.5

    for (int hd = 0; hd < 2; ++hd) {
      float qh[32];
      {
        const v8bf* q8 = (const v8bf*)(qkv + (long)s * NQKV + f * 64 + hd * 32);
        #pragma unroll
        for (int j = 0; j < 4; ++j) {
          v8bf qv = q8[j];
          #pragma unroll
          for (int e = 0; e < 8; ++e) qh[j * 8 + e] = bf2f(qv[e]);
        }
      }
      float lgt[9]; int np[9];
      #pragma unroll
      for (int win = 0; win < 9; ++win) {
        const int di = win / 3 - 1, dj = win % 3 - 1;
        const int hh = hs + di * dil, ww = wsrc + dj * dil;
        const bool ok = (hh >= 0) & (hh < W_) & (ww >= 0) & (ww < W_);
        np[win] = ok ? (bs * HW + hh * W_ + ww) : -1;
        float d = 0.0f;
        if (ok) {
          const v8bf* k8 = (const v8bf*)(qkv + (long)np[win] * NQKV + CIN + f * 64 + hd * 32);
          #pragma unroll
          for (int j = 0; j < 4; ++j) {
            v8bf kv8 = k8[j];
            #pragma unroll
            for (int e = 0; e < 8; ++e) d += qh[j * 8 + e] * bf2f(kv8[e]);
          }
        }
        lgt[win] = d * scale;                      // zero-padded windows -> 0
      }
      float mx = lgt[0];
      #pragma unroll
      for (int w2 = 1; w2 < 9; ++w2) mx = fmaxf(mx, lgt[w2]);
      float sum = 0.0f;
      #pragma unroll
      for (int w2 = 0; w2 < 9; ++w2) { lgt[w2] = __expf(lgt[w2] - mx); sum += lgt[w2]; }
      const float inv = 1.0f / sum;

      float oh[32];
      #pragma unroll
      for (int c = 0; c < 32; ++c) oh[c] = 0.0f;
      #pragma unroll
      for (int win = 0; win < 9; ++win) {
        if (np[win] >= 0) {
          const v8bf* v8p = (const v8bf*)(qkv + (long)np[win] * NQKV + 2 * CIN + f * 64 + hd * 32);
          const float a = lgt[win] * inv;
          #pragma unroll
          for (int j = 0; j < 4; ++j) {
            v8bf vv = v8p[j];
            #pragma unroll
            for (int e = 0; e < 8; ++e) oh[j * 8 + e] += a * bf2f(vv[e]);
          }
        }
      }
      const int cb = half * 64 + hd * 32;
      #pragma unroll
      for (int c = 0; c < 32; ++c)
        sXR[pix][cb + c] = fmaxf(oh[c], 0.0f) + x[xres + (long)(cb + c) * HW];
    }
  }
  __syncthreads();

  // -------- phase 1b: LayerNorm stats, one pixel per thread --------
  if (t < 64) {
    float mu = 0.0f;
    for (int c = 0; c < CIN; ++c) mu += sXR[t][c];
    mu *= (1.0f / CIN);
    float var = 0.0f;
    for (int c = 0; c < CIN; ++c) { float d = sXR[t][c] - mu; var += d * d; }
    var *= (1.0f / CIN);
    sMu[t] = mu;
    sRs[t] = rsqrtf(var + 1e-5f);
  }
  __syncthreads();

  // -------- phase 1c: normalize once -> bf16 A-fragment layout --------
  {
    const int pix = t >> 1, half = t & 1;
    const float mu = sMu[pix], rs = sRs[pix];
    const int mt = pix >> 4;
    #pragma unroll
    for (int j = 0; j < 64; ++j) {
      const int c = half * 64 + j;
      const float xn = (sXR[pix][c] - mu) * rs * ln_g[c] + ln_b[c];
      int g, h;
      fragpos(c & 31, g, h);
      sXN[c >> 5][mt][(g << 4) | (pix & 15)][h] = f2bf(xn);
    }
  }

  // -------- phase 2: proj GEMM (64 x 128 x 128) via WMMA --------
  const int wv = t >> 5, ln = t & 31;
  const int lm = ln & 15, lg = ln >> 4;

  v8f zero = {};
  v8f acc[8];
  #pragma unroll
  for (int i = 0; i < 8; ++i) acc[i] = zero;

  for (int ks = 0; ks < 4; ++ks) {
    const int k0 = ks * 32;
    __syncthreads();
    for (int idx = t; idx < 32 * CIN; idx += 128) {   // stage proj_w chunk
      int kk = idx & 31, n = idx >> 5, g, h;
      fragpos(kk, g, h);
      sPWf[n >> 4][(g << 4) | (n & 15)][h] = f2bf(proj_w[n * CIN + (k0 + kk)]);
    }
    __syncthreads();

    #pragma unroll
    for (int i = 0; i < 8; ++i) {
      const int tt = wv * 8 + i;              // 32 tiles: 4 M x 8 N
      const int mt = tt >> 3, nt = tt & 7;
      v16bf A  = *(const v16bf*)&sXN[ks][mt][ln][0];
      v16bf Bf = *(const v16bf*)&sPWf[nt][ln][0];
      acc[i] = __builtin_amdgcn_wmma_f32_16x16x32_bf16(
          false, A, false, Bf, (short)0, acc[i], false, false);
    }
  }

  // -------- epilogue: +proj_b + xr residual, 2x float4 per lane --------
  #pragma unroll
  for (int i = 0; i < 8; ++i) {
    const int tt = wv * 8 + i;
    const int mt = tt >> 3, nt = tt & 7;
    const int n = nt * 16 + lm;
    const float pb = proj_b[n];
    const int m0  = mt * 16 + (lg << 3);
    const int pp0 = p0 + m0;
    const int bp = pp0 / HW, qp0 = pp0 % HW;
    float* dst = out + (long)bp * CIN * HW + (long)n * HW + qp0;
    v4f o0, o1;
    #pragma unroll
    for (int r = 0; r < 4; ++r) o0[r] = acc[i][r]     + pb + sXR[m0 + r][n];
    #pragma unroll
    for (int r = 0; r < 4; ++r) o1[r] = acc[i][4 + r] + pb + sXR[m0 + 4 + r][n];
    *(v4f*)dst       = o0;
    *((v4f*)dst + 1) = o1;
  }
}

// ---------------------------------------------------------------------------
extern "C" void kernel_launch(void* const* d_in, const int* in_sizes, int n_in,
                              void* d_out, int out_size, void* d_ws, size_t ws_size,
                              hipStream_t stream) {
  const float* x      = (const float*)d_in[0];
  const float* hx     = (const float*)d_in[1];
  const float* q_w    = (const float*)d_in[2];
  const float* q_b    = (const float*)d_in[3];
  const float* kv_w   = (const float*)d_in[4];
  const float* ln_g   = (const float*)d_in[5];
  const float* ln_b   = (const float*)d_in[6];
  const float* proj_w = (const float*)d_in[7];
  const float* proj_b = (const float*)d_in[8];
  float* out = (float*)d_out;

  __bf16* qkv = (__bf16*)d_ws;     // B*HW*384 bf16 = 108 MB (fits in L2)

  const int npix = BATCH * HW;     // 147456
  qkv_kernel<<<npix / 32, 256, 0, stream>>>(x, hx, q_w, q_b, kv_w, qkv);
  attn_ln_proj_kernel<<<npix / 64, 128, 0, stream>>>(qkv, x, ln_g, ln_b,
                                                     proj_w, proj_b, out);
}
